// ContextualCircuit_17652315587197
// MI455X (gfx1250) — matrix operand order repaired and checked by reading, not compiled, not run on hardware
//
#include <hip/hip_runtime.h>

// ---------------------------------------------------------------------------
// ContextualCircuit on MI455X (gfx1250, wave32)
//   K=32 channels, 4x128x128 images, 100 recurrent steps.
//   Depthwise ring convs reduced to separable box sums (exact: kernels are
//   normalized box-minus-center / box-minus-inner-block, identical per chan).
//   1x1 channel mixes done with v_wmma_f32_16x16x32_bf16 (f32 accumulate).
//   Row staging uses CDNA5 global->LDS async copies (ASYNCcnt path).
// ---------------------------------------------------------------------------

typedef __attribute__((ext_vector_type(16))) __bf16 v16bf;
typedef __attribute__((ext_vector_type(8)))  float  v8f;

#define KCH      32
#define IMG_W    128
#define IMG_H    128
#define IMG_PIX  (IMG_W * IMG_H)          // 16384
#define NPLANES  (4 * KCH)                // 128 planes (img*chan)
#define NTOT     ((size_t)4 * KCH * IMG_PIX)  // 2,097,152 elements

// LDS byte offset of a __shared__ object: flat shared pointers carry the
// workgroup-relative LDS offset in their low 32 bits (ISA: LDS aperture
// truncates addr[31:0]).
__device__ __forceinline__ unsigned lds_off(const void* p) {
    return (unsigned)(uintptr_t)p;
}

// CDNA5 async global->LDS copy, one b32 per lane. Tracked by ASYNCcnt.
__device__ __forceinline__ void async_g2l_b32(unsigned lds_byte, const float* gptr) {
    unsigned long long ga = (unsigned long long)(uintptr_t)gptr;
    asm volatile("global_load_async_to_lds_b32 %0, %1, off"
                 :: "v"(lds_byte), "v"(ga) : "memory");
}

__device__ __forceinline__ void wait_asynccnt0() {
    asm volatile("s_wait_asynccnt 0x0" ::: "memory");
}

// -------------------- pass 1: horizontal sliding sums ----------------------
// grid: NPLANES*IMG_H blocks of 128 threads; one block = one row of one plane
__global__ __launch_bounds__(128) void cc_horiz(
    const float* __restrict__ Xin, const float* __restrict__ Yin,
    float* __restrict__ h9x, float* __restrict__ h9y, float* __restrict__ h29y)
{
    const int row   = blockIdx.x & (IMG_H - 1);
    const int plane = blockIdx.x >> 7;
    const size_t base = (size_t)plane * IMG_PIX + (size_t)row * IMG_W;
    const int t = threadIdx.x;

    __shared__ float rx[IMG_W + 28];
    __shared__ float ry[IMG_W + 28];

    // Direct global->LDS async fill of the row bodies (no VGPR round-trip).
    async_g2l_b32(lds_off(&rx[14 + t]), Xin + base + t);
    async_g2l_b32(lds_off(&ry[14 + t]), Yin + base + t);

    // Zero halos with normal ds stores while async copies are in flight.
    if (t < 28) {                       // zero halo (SAME zero padding)
        int hi = (t < 14) ? t : (IMG_W + 14 + (t - 14));
        rx[hi] = 0.0f; ry[hi] = 0.0f;
    }
    wait_asynccnt0();                   // this wave's async writes landed
    __syncthreads();                    // all waves' writes visible

    float s9x = 0.0f, s9y = 0.0f, s29 = 0.0f;
#pragma unroll
    for (int d = -4; d <= 4; ++d) {
        s9x += rx[14 + t + d];
        s9y += ry[14 + t + d];
    }
#pragma unroll
    for (int d = -14; d <= 14; ++d) s29 += ry[14 + t + d];

    h9x [base + t] = s9x;
    h9y [base + t] = s9y;
    h29y[base + t] = s29;
}

// -------------------- pass 2: vertical sums -> T, P surrounds --------------
// grid: NPLANES*IMG_H blocks of 128 threads (threads along j => coalesced)
__global__ __launch_bounds__(128) void cc_vert(
    const float* __restrict__ Xin,
    const float* __restrict__ h9x, const float* __restrict__ h9y,
    const float* __restrict__ h29y,
    float* __restrict__ Tbuf, float* __restrict__ Pbuf)
{
    const int row   = blockIdx.x & (IMG_H - 1);
    const int plane = blockIdx.x >> 7;
    const int j     = threadIdx.x;
    const size_t pbase = (size_t)plane * IMG_PIX;

    float v9x = 0.0f, v9y = 0.0f, v29 = 0.0f;
#pragma unroll
    for (int d = -4; d <= 4; ++d) {
        int r = row + d;
        if (r >= 0 && r < IMG_H) {
            v9x += h9x[pbase + (size_t)r * IMG_W + j];
            v9y += h9y[pbase + (size_t)r * IMG_W + j];
        }
    }
#pragma unroll
    for (int d = -14; d <= 14; ++d) {
        int r = row + d;
        if (r >= 0 && r < IMG_H) v29 += h29y[pbase + (size_t)r * IMG_W + j];
    }
    const size_t g = pbase + (size_t)row * IMG_W + j;
    Tbuf[g] = (v29 - v9y) * (1.0f / 760.0f);          // far ring (ex-bias)
    Pbuf[g] = (v9x - Xin[g]) * (1.0f / 80.0f);        // near ring (ex-bias)
}

// -------------------- pass 3: WMMA channel mixes + state update ------------
// 512 blocks x 256 threads (8 waves). Each wave: one 32chan x 16pix tile.
__global__ __launch_bounds__(256) void cc_mixupdate(
    const float* __restrict__ Xin, const float* __restrict__ Yin,
    const float* __restrict__ inbound,
    const float* __restrict__ Tbuf, const float* __restrict__ Pbuf,
    const float* __restrict__ Wu, const float* __restrict__ Wb,
    const float* __restrict__ bu, const float* __restrict__ bb,
    const float* __restrict__ bn, const float* __restrict__ bf,
    float* __restrict__ Xout, float* __restrict__ Yout)
{
    __shared__ __bf16 sWu[KCH * KCH];
    __shared__ __bf16 sWb[KCH * KCH];
    __shared__ float  sbu[KCH], sbb[KCH], sbn[KCH], sbf[KCH];

    const int t = threadIdx.x;
    for (int i = t; i < KCH * KCH; i += 256) {
        sWu[i] = (__bf16)Wu[i];
        sWb[i] = (__bf16)Wb[i];
    }
    if (t < KCH) { sbu[t] = bu[t]; sbb[t] = bb[t]; sbn[t] = bn[t]; sbf[t] = bf[t]; }
    __syncthreads();

    const int wave = t >> 5;
    const int lane = t & 31;
    const int tile = blockIdx.x * 8 + wave;     // 0..4095
    const int img  = tile >> 10;                // 1024 tiles per image
    const int pix0 = (tile & 1023) << 4;        // 16 pixels per tile
    const int h    = lane >> 4;                 // lane half
    const int n    = lane & 15;                 // pixel within tile / A-row
    const int pix  = pix0 + n;

    const size_t ibase = (size_t)img * KCH * IMG_PIX + pix;
    const float* Xb = Xin + ibase;
    const float* Yb = Yin + ibase;

    // ---- B operands: 32x16 bf16, lanes 0-15 K=0..15, lanes 16-31 K=16..31
    v16bf bY, bX;
#pragma unroll
    for (int v = 0; v < 8; ++v) {
        const int k0 = 16 * h + 2 * v;
        bY[2 * v + 0] = (__bf16)Yb[(size_t)(k0 + 0) * IMG_PIX];
        bY[2 * v + 1] = (__bf16)Yb[(size_t)(k0 + 1) * IMG_PIX];
        bX[2 * v + 0] = (__bf16)Xb[(size_t)(k0 + 0) * IMG_PIX];
        bX[2 * v + 1] = (__bf16)Xb[(size_t)(k0 + 1) * IMG_PIX];
    }

    // ---- A operands: 16x32 bf16 (VGPR v: K=16h+2v, matching B);
    //      two row blocks (out chans 0-15, 16-31)
    v16bf aU0, aU1, aQ0, aQ1;
#pragma unroll
    for (int v = 0; v < 8; ++v) {
        const int kb = 16 * h + 2 * v;
        aU0[2 * v + 0] = sWu[n * KCH + kb];
        aU0[2 * v + 1] = sWu[n * KCH + kb + 1];
        aU1[2 * v + 0] = sWu[(16 + n) * KCH + kb];
        aU1[2 * v + 1] = sWu[(16 + n) * KCH + kb + 1];
        aQ0[2 * v + 0] = sWb[n * KCH + kb];
        aQ0[2 * v + 1] = sWb[n * KCH + kb + 1];
        aQ1[2 * v + 0] = sWb[(16 + n) * KCH + kb];
        aQ1[2 * v + 1] = sWb[(16 + n) * KCH + kb + 1];
    }

    const v8f z = {0.f, 0.f, 0.f, 0.f, 0.f, 0.f, 0.f, 0.f};
    v8f u0 = __builtin_amdgcn_wmma_f32_16x16x32_bf16(false, aU0, false, bY, (short)0, z, false, false);
    v8f u1 = __builtin_amdgcn_wmma_f32_16x16x32_bf16(false, aU1, false, bY, (short)0, z, false, false);
    v8f q0 = __builtin_amdgcn_wmma_f32_16x16x32_bf16(false, aQ0, false, bX, (short)0, z, false, false);
    v8f q1 = __builtin_amdgcn_wmma_f32_16x16x32_bf16(false, aQ1, false, bX, (short)0, z, false, false);

    // ---- elementwise update. D layout: VGPR r <-> out-chan m = blk*16+8h+r,
    //      pixel = pix (lane's n). Constants: dt=0.5, 1-k=0.875, FF=4.5,
    //      UI: X+1, TI: 3X+0.3, FB=TE=BE=1.
    const float* ib = inbound + ibase;
#pragma unroll
    for (int blk = 0; blk < 2; ++blk) {
#pragma unroll
        for (int r = 0; r < 8; ++r) {
            const int m = blk * 16 + 8 * h + r;
            const size_t off = (size_t)m * IMG_PIX;
            const size_t g = ibase + off;
            const float mixU = blk ? u1[r] : u0[r];
            const float mixQ = blk ? q1[r] : q0[r];
            const float X = Xb[off];
            const float Y = Yb[off];
            const float U = mixU + sbu[m];
            const float Q = mixQ + sbb[m];
            const float T = Tbuf[g] + sbf[m];
            const float P = Pbuf[g] + sbn[m];
            const float cX = fmaxf(4.5f * ib[off] - (X + 1.0f) * U - (3.0f * X + 0.3f) * T, 0.0f);
            const float cY = fmaxf(X + P + Q, 0.0f);
            Xout[g] = 0.875f * X + 0.5f * cX;
            Yout[g] = 0.875f * Y + 0.5f * cY;
        }
    }
}

// ---------------------------------------------------------------------------
extern "C" void kernel_launch(void* const* d_in, const int* in_sizes, int n_in,
                              void* d_out, int out_size, void* d_ws, size_t ws_size,
                              hipStream_t stream)
{
    (void)in_sizes; (void)n_in; (void)out_size; (void)ws_size;

    const float* inbound = (const float*)d_in[0];
    const float* Wu = (const float*)d_in[1];   // w_untuned (32,32)
    const float* bu = (const float*)d_in[2];   // b_untuned
    const float* Wb = (const float*)d_in[3];   // w_broad   (32,32)
    const float* bb = (const float*)d_in[4];   // b_broad
    // d_in[5] = w_near (structure folded into box sums), d_in[7] = w_far
    const float* bn = (const float*)d_in[6];   // b_near
    const float* bf = (const float*)d_in[8];   // b_far

    float* ws = (float*)d_ws;                  // needs 9 * 8 MB = 72 MB
    float* Xbuf[2] = { ws + 0 * NTOT, ws + 1 * NTOT };
    float* Ybuf[2] = { ws + 2 * NTOT, ws + 3 * NTOT };
    float* h9x  = ws + 4 * NTOT;
    float* h9y  = ws + 5 * NTOT;
    float* h29y = ws + 6 * NTOT;
    float* Tbuf = ws + 7 * NTOT;
    float* Pbuf = ws + 8 * NTOT;

    const dim3 gHV(NPLANES * IMG_H);           // 16384 row-blocks
    const dim3 bHV(IMG_W);                     // 128 threads
    const dim3 gMU(512);                       // 512 * 8 waves * 16 pix = 65536
    const dim3 bMU(256);

    const int NUM_ITERS = 100;
    for (int it = 0; it < NUM_ITERS; ++it) {
        const float* Xin = (it == 0) ? inbound : Xbuf[(it + 1) & 1];
        const float* Yin = (it == 0) ? inbound : Ybuf[(it + 1) & 1];
        float* Xout = Xbuf[it & 1];
        float* Yout = (it == NUM_ITERS - 1) ? (float*)d_out : Ybuf[it & 1];

        cc_horiz<<<gHV, bHV, 0, stream>>>(Xin, Yin, h9x, h9y, h29y);
        cc_vert <<<gHV, bHV, 0, stream>>>(Xin, h9x, h9y, h29y, Tbuf, Pbuf);
        cc_mixupdate<<<gMU, bMU, 0, stream>>>(Xin, Yin, inbound, Tbuf, Pbuf,
                                              Wu, Wb, bu, bb, bn, bf,
                                              Xout, Yout);
    }
}
